// TextGenerationModel_44040594653800
// MI455X (gfx1250) — matrix operand order repaired and checked by reading, not compiled, not run on hardware
//
#include <hip/hip_runtime.h>
#include <cstddef>

// ---------------- model dims (match reference) ----------------
static constexpr int V  = 32000, E = 256, H = 512;
static constexpr int PE = 64,  PH = 128;
static constexpr int B  = 32,  T = 64, TP = 12;
static constexpr int NT  = B * T;          // 2048 word positions
static constexpr int RR  = NT * TP;        // 24576 phoneme rows
static constexpr int DIN = E + PH;         // 384 main-LSTM input dim
static constexpr int G4H = 4 * H;          // 2048 gate cols (main)
static constexpr int G4P = 4 * PH;         // 512 gate cols (phoneme)

// ---------------- WMMA fragment types ----------------
typedef __attribute__((ext_vector_type(16))) __bf16        v16bf;
typedef __attribute__((ext_vector_type(8)))  float         v8f;
typedef __attribute__((ext_vector_type(4)))  unsigned int  v4u;

union Frag { v4u q[2]; v16bf v; };

#define WMMA_BF16(a, b, c) \
  __builtin_amdgcn_wmma_f32_16x16x32_bf16(false, (a), false, (b), (short)0, (c), false, false)

__device__ __forceinline__ v8f v8f_zero() {
  v8f z;
#pragma unroll
  for (int i = 0; i < 8; ++i) z[i] = 0.0f;
  return z;
}

__device__ __forceinline__ unsigned short f2bf_bits(float x) {
  union { float f; unsigned u; } v; v.f = x;
  unsigned r = v.u + 0x7FFFu + ((v.u >> 16) & 1u);   // RNE
  return (unsigned short)(r >> 16);
}

__device__ __forceinline__ float sigf(float x) { return 1.0f / (1.0f + __expf(-x)); }

// A-matrix 16x32 bf16 per ISA layout: lane<16 -> K {0..7,16..23}; lane>=16 -> K {8..15,24..31}.
// base already includes row*lda + (lane<16 ? 0 : 8).
__device__ __forceinline__ v16bf load_a(const unsigned short* base, int k0) {
  Frag f; const v4u* p = (const v4u*)(base + k0);
  f.q[0] = p[0];   // k = kb+0..7
  f.q[1] = p[2];   // k = kb+16..23
  return f.v;
}
// B-matrix 32x16 bf16: lane<16 -> K {0..7,16..23}, lane>=16 -> K {8..15,24..31} (col = lane&15).
// base already includes n_row*ldb + (lane<16 ? 0 : 8).
__device__ __forceinline__ v16bf load_b(const unsigned short* base, int k0) {
  Frag f; const v4u* p = (const v4u*)(base + k0);
  f.q[0] = p[0];
  f.q[1] = p[2];
  return f.v;
}

// ---------------- utility kernels ----------------
__global__ void cvt_f32_bf16(const float* __restrict__ src, unsigned short* __restrict__ dst, int n) {
  int i = blockIdx.x * blockDim.x + threadIdx.x;
  if (i < n) dst[i] = f2bf_bits(src[i]);
}

__global__ void zero_u32(unsigned int* __restrict__ p, int ndw) {
  int i = blockIdx.x * blockDim.x + threadIdx.x;
  if (i < ndw) p[i] = 0u;
}

// word embedding -> bf16 into concat-input buffer cols [0,256)
__global__ void embed_words(const int* __restrict__ x, const float* __restrict__ emb,
                            unsigned short* __restrict__ inputs) {
  int i = blockIdx.x * blockDim.x + threadIdx.x;   // over NT*E
  if (i >= NT * E) return;
  int n = i >> 8, e = i & 255;
  inputs[(size_t)n * DIN + e] = f2bf_bits(emb[(size_t)x[n] * E + e]);
}

// phoneme embedding gather -> bf16 A matrix [RR, PE]
__global__ void embed_ph(const int* __restrict__ xp, const float* __restrict__ pemb,
                         unsigned short* __restrict__ phx) {
  int i = blockIdx.x * blockDim.x + threadIdx.x;   // over RR*PE
  if (i >= RR * PE) return;
  int r = i >> 6, e = i & 63;
  phx[(size_t)r * PE + e] = f2bf_bits(pemb[(size_t)xp[r] * PE + e]);
}

// ---------------- generic bf16 WMMA GEMM: C[M,N] = A[M,K] * Bw[N,K]^T + bias0 + bias1 ----------------
__global__ void gemm_bf16(const unsigned short* __restrict__ A, int lda,
                          const unsigned short* __restrict__ Bw, int ldb,
                          float* __restrict__ C, int ldc,
                          int M, int N, int K,
                          const float* __restrict__ bias0, const float* __restrict__ bias1) {
  int tid  = blockIdx.x * blockDim.x + threadIdx.x;
  int wave = tid >> 5;
  int lane = threadIdx.x & 31;
  int ntiles = N >> 4;
  int total  = (M >> 4) * ntiles;
  if (wave >= total) return;                 // wave-uniform: EXEC stays all-ones for WMMA
  int mt = wave / ntiles, nt = wave - mt * ntiles;
  int m0 = mt << 4, n0 = nt << 4;
  int ln = lane & 15;
  const unsigned short* arow = A  + (size_t)(m0 + ln) * lda + ((lane < 16) ? 0 : 8);
  const unsigned short* brow = Bw + (size_t)(n0 + ln) * ldb + ((lane < 16) ? 0 : 8);

  v8f acc = v8f_zero();
  for (int k0 = 0; k0 < K; k0 += 32) {
    v16bf a = load_a(arow, k0);
    v16bf b = load_b(brow, k0);
    acc = WMMA_BF16(a, b, acc);
  }
  int col = n0 + ln;
  float bias = 0.0f;
  if (bias0) bias += bias0[col];
  if (bias1) bias += bias1[col];
#pragma unroll
  for (int r = 0; r < 8; ++r) {
    int row = m0 + r + ((lane < 16) ? 0 : 8);
    C[(size_t)row * ldc + col] = acc[r] + bias;
  }
}

// ---------------- phoneme LSTM step (batch 2048, hidden 128) ----------------
// block = 256 thr (8 waves), grid = 128 (16 rows each). wave w owns hidden cols [w*16, w*16+16):
// computes the 4 gate tiles at cols {j, 128+j, 256+j, 384+j} so i/f/g/o recombine in-register.
__global__ void ph_step(const float* __restrict__ xgp,            // [RR, 512] precomputed input gates
                        const unsigned short* __restrict__ Wb,    // bf16 Whh_p [512,128]
                        const unsigned short* __restrict__ h_prev,// bf16 [2048,128]
                        unsigned short* __restrict__ h_next,      // bf16 [2048,128]
                        float* __restrict__ c_ws,                 // f32  [2048,128]
                        const int* __restrict__ plens,            // [2048]
                        unsigned short* __restrict__ inputs,      // bf16 [2048,384]; cols 256.. get last-h
                        int t) {
  int lane = threadIdx.x & 31;
  int wave = threadIdx.x >> 5;
  int m0 = blockIdx.x << 4;
  int ln = lane & 15;
  int j0 = wave << 4;
  const unsigned short* arow = h_prev + (size_t)(m0 + ln) * PH + ((lane < 16) ? 0 : 8);

  v8f acc[4];
#pragma unroll
  for (int g = 0; g < 4; ++g) acc[g] = v8f_zero();

#pragma unroll
  for (int ks = 0; ks < 4; ++ks) {           // K = 128
    v16bf a = load_a(arow, ks * 32);
#pragma unroll
    for (int g = 0; g < 4; ++g) {
      int n0 = g * PH + j0;
      const unsigned short* brow = Wb + (size_t)(n0 + ln) * PH + ((lane < 16) ? 0 : 8);
      v16bf b = load_b(brow, ks * 32);
      acc[g] = WMMA_BF16(a, b, acc[g]);
    }
  }

  int col = j0 + ln;                          // hidden index 0..127
#pragma unroll
  for (int r = 0; r < 8; ++r) {
    int n = m0 + r + ((lane < 16) ? 0 : 8);
    const float* xrow = xgp + ((size_t)n * TP + t) * G4P;
    float gi = acc[0][r] + xrow[col];
    float gf = acc[1][r] + xrow[PH + col];
    float gg = acc[2][r] + xrow[2 * PH + col];
    float go = acc[3][r] + xrow[3 * PH + col];
    float c  = sigf(gf) * c_ws[(size_t)n * PH + col] + sigf(gi) * tanhf(gg);
    c_ws[(size_t)n * PH + col] = c;
    unsigned short hb = f2bf_bits(sigf(go) * tanhf(c));
    h_next[(size_t)n * PH + col] = hb;
    int pl = plens[n];
    int lt = (pl > 1) ? (pl - 1) : 0;
    if (t == lt) inputs[(size_t)n * DIN + E + col] = hb;   // fused "gather last valid step"
  }
}

// ---------------- main LSTM step (batch 32, hidden 512) ----------------
// grid = 8 blocks x 256 thr: blk>>2 -> m-tile (batch rows 0-15 / 16-31), (blk&3,wave) -> 16-col j slice.
__global__ void main_step(const float* __restrict__ xg,            // [NT, 2048] precomputed gates
                          const unsigned short* __restrict__ Wb,   // bf16 Whh [2048,512]
                          const unsigned short* __restrict__ h_prev,// bf16 [32,512]
                          unsigned short* __restrict__ h_next,     // bf16 [32,512]
                          float* __restrict__ c_ws,                // f32  [32,512]
                          unsigned short* __restrict__ hs,         // bf16 [NT,512] all h outputs
                          int t) {
  int lane = threadIdx.x & 31;
  int wave = threadIdx.x >> 5;
  int m0 = (blockIdx.x >> 2) << 4;
  int jt = ((blockIdx.x & 3) << 3) + wave;
  int j0 = jt << 4;
  int ln = lane & 15;
  const unsigned short* arow = h_prev + (size_t)(m0 + ln) * H + ((lane < 16) ? 0 : 8);

  v8f acc[4];
#pragma unroll
  for (int g = 0; g < 4; ++g) acc[g] = v8f_zero();

  for (int ks = 0; ks < 16; ++ks) {          // K = 512
    v16bf a = load_a(arow, ks * 32);
#pragma unroll
    for (int g = 0; g < 4; ++g) {
      int n0 = g * H + j0;
      const unsigned short* brow = Wb + (size_t)(n0 + ln) * H + ((lane < 16) ? 0 : 8);
      v16bf b = load_b(brow, ks * 32);
      acc[g] = WMMA_BF16(a, b, acc[g]);
    }
  }

  int col = j0 + ln;                          // hidden index 0..511
#pragma unroll
  for (int r = 0; r < 8; ++r) {
    int bidx = m0 + r + ((lane < 16) ? 0 : 8);      // batch 0..31
    const float* xrow = xg + ((size_t)bidx * T + t) * G4H;
    float gi = acc[0][r] + xrow[col];
    float gf = acc[1][r] + xrow[H + col];
    float gg = acc[2][r] + xrow[2 * H + col];
    float go = acc[3][r] + xrow[3 * H + col];
    float c  = sigf(gf) * c_ws[(size_t)bidx * H + col] + sigf(gi) * tanhf(gg);
    c_ws[(size_t)bidx * H + col] = c;
    unsigned short hb = f2bf_bits(sigf(go) * tanhf(c));
    h_next[(size_t)bidx * H + col] = hb;
    hs[((size_t)bidx * T + t) * H + col] = hb;
  }
}

// ---------------- host-side orchestration ----------------
extern "C" void kernel_launch(void* const* d_in, const int* in_sizes, int n_in,
                              void* d_out, int out_size, void* d_ws, size_t ws_size,
                              hipStream_t stream) {
  (void)in_sizes; (void)n_in; (void)out_size; (void)ws_size;

  const int*   x    = (const int*)d_in[0];
  const int*   xp   = (const int*)d_in[1];
  /* d_in[2] = lengths: unused by the reference */
  const int*   plen = (const int*)d_in[3];
  const float* emb  = (const float*)d_in[4];
  const float* pemb = (const float*)d_in[5];
  const float* WihP = (const float*)d_in[6];
  const float* WhhP = (const float*)d_in[7];
  const float* bihP = (const float*)d_in[8];
  const float* bhhP = (const float*)d_in[9];
  const float* Wih  = (const float*)d_in[10];
  const float* Whh  = (const float*)d_in[11];
  const float* bih  = (const float*)d_in[12];
  const float* bhh  = (const float*)d_in[13];
  const float* fcW  = (const float*)d_in[14];
  const float* fcb  = (const float*)d_in[15];
  float* out = (float*)d_out;

  // bump allocator over d_ws
  char* wp = (char*)d_ws;
  auto alloc = [&](size_t bytes) -> void* {
    void* r = (void*)wp;
    wp += (bytes + 255) & ~(size_t)255;
    return r;
  };
  unsigned short* wWihP = (unsigned short*)alloc((size_t)G4P * PE  * 2);
  unsigned short* wWhhP = (unsigned short*)alloc((size_t)G4P * PH  * 2);
  unsigned short* wWih  = (unsigned short*)alloc((size_t)G4H * DIN * 2);
  unsigned short* wWhh  = (unsigned short*)alloc((size_t)G4H * H   * 2);
  unsigned short* wFc   = (unsigned short*)alloc((size_t)V   * H   * 2);
  unsigned short* phx   = (unsigned short*)alloc((size_t)RR  * PE  * 2);
  unsigned short* inp   = (unsigned short*)alloc((size_t)NT  * DIN * 2);
  unsigned short* hp0   = (unsigned short*)alloc((size_t)NT  * PH  * 2);
  unsigned short* hp1   = (unsigned short*)alloc((size_t)NT  * PH  * 2);
  unsigned short* h0    = (unsigned short*)alloc((size_t)B   * H   * 2);
  unsigned short* h1    = (unsigned short*)alloc((size_t)B   * H   * 2);
  unsigned short* hs    = (unsigned short*)alloc((size_t)NT  * H   * 2);
  float* xgp = (float*)alloc((size_t)RR * G4P * 4);
  float* xg  = (float*)alloc((size_t)NT * G4H * 4);
  float* cp  = (float*)alloc((size_t)NT * PH  * 4);
  float* cm  = (float*)alloc((size_t)B  * H   * 4);

  auto cvt = [&](const float* s, unsigned short* d, int n) {
    cvt_f32_bf16<<<(n + 255) / 256, 256, 0, stream>>>(s, d, n);
  };
  auto zero = [&](void* p, int ndw) {
    zero_u32<<<(ndw + 255) / 256, 256, 0, stream>>>((unsigned int*)p, ndw);
  };
  auto gemm = [&](const unsigned short* A, int lda, const unsigned short* Bm, int ldb,
                  float* C, int ldc, int M, int N, int K,
                  const float* b0, const float* b1) {
    int waves = (M >> 4) * (N >> 4);
    gemm_bf16<<<(waves + 7) / 8, 256, 0, stream>>>(A, lda, Bm, ldb, C, ldc, M, N, K, b0, b1);
  };

  // 1) weights -> bf16 (deterministic, redone every call)
  cvt(WihP, wWihP, G4P * PE);
  cvt(WhhP, wWhhP, G4P * PH);
  cvt(Wih,  wWih,  G4H * DIN);
  cvt(Whh,  wWhh,  G4H * H);
  cvt(fcW,  wFc,   V * H);

  // 2) zero recurrent state (h ping buffers read at t=0, c buffers)
  zero(hp0, NT * PH / 2);
  zero(h0,  B * H / 2);
  zero(cp,  NT * PH);
  zero(cm,  B * H);

  // 3) phoneme embedding gather -> bf16 [RR, PE]
  embed_ph<<<(RR * PE) / 256, 256, 0, stream>>>(xp, pemb, phx);

  // 4) phoneme xg = phx @ Wih_p^T + bih_p + bhh_p   [RR, 512]
  gemm(phx, PE, wWihP, PE, xgp, G4P, RR, G4P, PE, bihP, bhhP);

  // 5) phoneme LSTM recurrence (12 steps); fuses last-valid-step gather into `inp` cols 256..383
  for (int t = 0; t < TP; ++t) {
    const unsigned short* hin  = (t & 1) ? hp1 : hp0;
    unsigned short*       hout = (t & 1) ? hp0 : hp1;
    ph_step<<<NT / 16, 256, 0, stream>>>(xgp, wWhhP, hin, hout, cp, plen, inp, t);
  }

  // 6) word embedding -> bf16 into `inp` cols 0..255
  embed_words<<<(NT * E) / 256, 256, 0, stream>>>(x, emb, inp);

  // 7) main xg = inp @ Wih^T + bih + bhh   [NT, 2048]
  gemm(inp, DIN, wWih, DIN, xg, G4H, NT, G4H, DIN, bih, bhh);

  // 8) main LSTM recurrence (64 steps), h outputs -> hs bf16 [NT, 512]
  for (int t = 0; t < T; ++t) {
    const unsigned short* hin  = (t & 1) ? h1 : h0;
    unsigned short*       hout = (t & 1) ? h0 : h1;
    main_step<<<8, 256, 0, stream>>>(xg, wWhh, hin, hout, cm, hs, t);
  }

  // 9) logits = hs @ fcW^T + fcb   [NT, 32000] -> d_out (store-bound: 262 MB fp32)
  gemm(hs, H, wFc, H, out, V, NT, V, H, fcb, nullptr);
}
